// InterEnsembleLearningTransformer_64183991271581
// MI455X (gfx1250) — matrix operand order repaired and checked by reading, not compiled, not run on hardware
//
#include <hip/hip_runtime.h>
#include <math.h>

// ---- model dims ----
#define BSZ   2
#define CDIM  768
#define NHEAD 12
#define DH    64
#define MLPD  3072
#define NTOK  197
#define PNUM  196
#define NCLS  200
#define NVOTE 24

typedef __attribute__((ext_vector_type(16))) _Float16 v16h;
typedef __attribute__((ext_vector_type(8)))  _Float16 v8h;
typedef __attribute__((ext_vector_type(4)))  _Float16 v4h;
typedef __attribute__((ext_vector_type(8)))  float    v8f;
typedef __attribute__((ext_vector_type(4)))  float    vf4;

// =====================================================================
// Generic WMMA GEMM:  C[z] = alpha * (A[z] @ W[z]) + bias + R[z]
//   A: [M,K] fp32, row stride lda (lda%4==0, bases 16B aligned — host
//      guarantees this; scores buffer uses a padded row stride)
//   W: wtrans=0 -> W[k*ldw + n],  wtrans=1 -> W[n*ldw + k]
// One wave computes a 16x32 C tile: one A fragment, two B fragments,
// two v_wmma_f32_16x16x32_f16 per K-step.
// LDS layouts give contiguous 128-bit DS fragment loads:
//   Asm[m][k] (16x32), Bsm[n][k] (32x32)
// =====================================================================

__device__ __forceinline__ void frag_mma(const _Float16* Asm, const _Float16* Bsm,
                                         int lane, v8f& acc0, v8f& acc1)
{
    const int g  = lane >> 4;
    const int ml = lane & 15;
    v8h alo = *(const v8h*)(Asm + ml * 32 + (g << 4));
    v8h ahi = *(const v8h*)(Asm + ml * 32 + (g << 4) + 8);
    v16h af = __builtin_shufflevector(alo, ahi, 0,1,2,3,4,5,6,7,8,9,10,11,12,13,14,15);

    v8h b0lo = *(const v8h*)(Bsm + ml * 32 + (g << 4));
    v8h b0hi = *(const v8h*)(Bsm + ml * 32 + (g << 4) + 8);
    v16h bf0 = __builtin_shufflevector(b0lo, b0hi, 0,1,2,3,4,5,6,7,8,9,10,11,12,13,14,15);

    v8h b1lo = *(const v8h*)(Bsm + (ml + 16) * 32 + (g << 4));
    v8h b1hi = *(const v8h*)(Bsm + (ml + 16) * 32 + (g << 4) + 8);
    v16h bf1 = __builtin_shufflevector(b1lo, b1hi, 0,1,2,3,4,5,6,7,8,9,10,11,12,13,14,15);

    acc0 = __builtin_amdgcn_wmma_f32_16x16x32_f16(false, af, false, bf0, (short)0, acc0, false, false);
    acc1 = __builtin_amdgcn_wmma_f32_16x16x32_f16(false, af, false, bf1, (short)0, acc1, false, false);
}

// ---- fast K-step: tile fully in bounds, vec4 global loads ----
__device__ __forceinline__ void kstep_fast(const float* __restrict__ A, int lda,
                                           const float* __restrict__ W, int ldw, int wtrans,
                                           int m0, int n0, int kt, int K,
                                           int lane, _Float16* Asm, _Float16* Bsm,
                                           v8f& acc0, v8f& acc1)
{
    __syncthreads();

    // A tile 16x32: 128 float4 chunks, 4 per lane (row = c/8, col4 = c%8)
    #pragma unroll
    for (int i = 0; i < 4; ++i) {
        int c = i * 32 + lane;
        int row = c >> 3, col4 = c & 7;
        vf4 f = *(const vf4*)(A + (long)(m0 + row) * lda + kt + col4 * 4);
        v4h hh; hh[0] = (_Float16)f[0]; hh[1] = (_Float16)f[1];
                hh[2] = (_Float16)f[2]; hh[3] = (_Float16)f[3];
        *(v4h*)(Asm + row * 32 + col4 * 4) = hh;
    }
    // speculative prefetch of next A k-slab (dropped on fault per ISA)
    __builtin_prefetch(&A[(long)(m0 + (lane & 15)) * lda + kt + 32], 0, 1);

    // B tile 32(k) x 32(n) -> LDS transposed [n][k]
    if (wtrans) {  // W[n*ldw + k]: contiguous along k
        #pragma unroll
        for (int i = 0; i < 8; ++i) {
            int c = i * 32 + lane;
            int n = c >> 3, col4 = c & 7;
            vf4 f = *(const vf4*)(W + (long)(n0 + n) * ldw + kt + col4 * 4);
            v4h hh; hh[0] = (_Float16)f[0]; hh[1] = (_Float16)f[1];
                    hh[2] = (_Float16)f[2]; hh[3] = (_Float16)f[3];
            *(v4h*)(Bsm + n * 32 + col4 * 4) = hh;
        }
    } else {       // W[k*ldw + n]: contiguous along n, scatter-transpose into LDS
        #pragma unroll
        for (int i = 0; i < 8; ++i) {
            int c = i * 32 + lane;
            int k = c >> 3, n4 = c & 7;
            vf4 f = *(const vf4*)(W + (long)(kt + k) * ldw + n0 + n4 * 4);
            #pragma unroll
            for (int j = 0; j < 4; ++j)
                Bsm[(n4 * 4 + j) * 32 + k] = (_Float16)f[j];
        }
    }
    __syncthreads();
    frag_mma(Asm, Bsm, lane, acc0, acc1);
}

// ---- guarded K-step: clamp addresses + mask (no branches) ----
template <bool GMN>
__device__ __forceinline__ void kstep_guard(const float* __restrict__ A, int lda,
                                            const float* __restrict__ W, int ldw, int wtrans,
                                            int m0, int n0, int kt, int M, int N, int K,
                                            int lane, _Float16* Asm, _Float16* Bsm,
                                            v8f& acc0, v8f& acc1)
{
    __syncthreads();

    #pragma unroll
    for (int i = 0; i < 16; ++i) {
        int gm = m0 + i, gk = kt + lane;
        int cm = GMN ? (gm < M ? gm : M - 1) : gm;
        int ck = gk < K ? gk : K - 1;
        float v = A[(long)cm * lda + ck];
        bool oob = (GMN && gm >= M) || (gk >= K);
        Asm[i * 32 + lane] = (_Float16)(oob ? 0.f : v);
    }
    if (wtrans) {
        #pragma unroll
        for (int i = 0; i < 32; ++i) {
            int gn = n0 + i, gk = kt + lane;
            int cn = GMN ? (gn < N ? gn : N - 1) : gn;
            int ck = gk < K ? gk : K - 1;
            float v = W[(long)cn * ldw + ck];
            bool oob = (GMN && gn >= N) || (gk >= K);
            Bsm[i * 32 + lane] = (_Float16)(oob ? 0.f : v);
        }
    } else {
        #pragma unroll
        for (int i = 0; i < 32; ++i) {
            int gk = kt + i, gn = n0 + lane;
            int ck = gk < K ? gk : K - 1;
            int cn = GMN ? (gn < N ? gn : N - 1) : gn;
            float v = W[(long)ck * ldw + cn];
            bool oob = (GMN && gn >= N) || (gk >= K);
            Bsm[lane * 32 + i] = (_Float16)(oob ? 0.f : v);
        }
    }
    __syncthreads();
    frag_mma(Asm, Bsm, lane, acc0, acc1);
}

__global__ __launch_bounds__(32)
void gemm16_wmma(const float* __restrict__ A, long sAb, long sAh, int lda,
                 const float* __restrict__ W, long sWb, long sWh, int ldw, int wtrans,
                 float* __restrict__ C, long sCb, long sCh, int ldc,
                 const float* __restrict__ bias,
                 const float* __restrict__ R, long sRb, int ldr,
                 int M, int N, int K, float alpha, int H)
{
    __shared__ __align__(16) _Float16 Asm[16 * 32];
    __shared__ __align__(16) _Float16 Bsm[32 * 32];

    int z = blockIdx.z;
    int b = z / H, h = z % H;
    A += (long)b * sAb + (long)h * sAh;
    W += (long)b * sWb + (long)h * sWh;
    C += (long)b * sCb + (long)h * sCh;
    if (R) R += (long)b * sRb;

    const int n0 = blockIdx.x * 32;
    const int m0 = blockIdx.y * 16;
    const int lane = threadIdx.x;

    v8f acc0 = {}, acc1 = {};

    const bool full = (m0 + 16 <= M) && (n0 + 32 <= N);
    const int kfull = K & ~31;

    if (full) {
        for (int kt = 0; kt < kfull; kt += 32)
            kstep_fast(A, lda, W, ldw, wtrans, m0, n0, kt, K, lane, Asm, Bsm, acc0, acc1);
        if (kfull < K)
            kstep_guard<false>(A, lda, W, ldw, wtrans, m0, n0, kfull, M, N, K, lane, Asm, Bsm, acc0, acc1);
    } else {
        for (int kt = 0; kt < K; kt += 32)
            kstep_guard<true>(A, lda, W, ldw, wtrans, m0, n0, kt, M, N, K, lane, Asm, Bsm, acc0, acc1);
    }

    // ---- epilogue: lane&15 = N, vgpr r -> M = r + (lane>>4)*8 ----
    const int g  = lane >> 4;
    const int nl = lane & 15;
    #pragma unroll
    for (int half = 0; half < 2; ++half) {
        int gn = n0 + nl + half * 16;
        v8f& acc = half ? acc1 : acc0;
        if (gn < N) {
            float bv = bias ? bias[gn] : 0.f;
            #pragma unroll
            for (int r = 0; r < 8; ++r) {
                int gm = m0 + r + g * 8;
                if (gm < M) {
                    float outv = acc[r] * alpha + bv;
                    if (R) outv += R[(long)gm * ldr + gn];
                    C[(long)gm * ldc + gn] = outv;
                }
            }
        }
    }
}

// ---------------- LayerNorm (one block per token row) ----------------
__global__ void ln_kernel(const float* __restrict__ x, long sXb,
                          const float* __restrict__ s, const float* __restrict__ bb,
                          float* __restrict__ y, long sYb)
{
    int b = blockIdx.y, n = blockIdx.x;
    const float* row = x + (long)b * sXb + (long)n * CDIM;
    float* out = y + (long)b * sYb + (long)n * CDIM;
    __shared__ float red[256];
    int t = threadIdx.x;

    float acc = 0.f;
    for (int c = t; c < CDIM; c += 256) acc += row[c];
    red[t] = acc; __syncthreads();
    for (int o = 128; o > 0; o >>= 1) { if (t < o) red[t] += red[t + o]; __syncthreads(); }
    float mean = red[0] / (float)CDIM;
    __syncthreads();

    acc = 0.f;
    for (int c = t; c < CDIM; c += 256) { float d = row[c] - mean; acc += d * d; }
    red[t] = acc; __syncthreads();
    for (int o = 128; o > 0; o >>= 1) { if (t < o) red[t] += red[t + o]; __syncthreads(); }
    float rstd = rsqrtf(red[0] / (float)CDIM + 1e-6f);

    for (int c = t; c < CDIM; c += 256)
        out[c] = (row[c] - mean) * rstd * s[c] + bb[c];
}

// ---------------- Softmax over rows of scores [B,H,Nq,(NtP)] ----------------
__global__ void softmax_kernel(float* __restrict__ sc, int Nt, int NtP)
{
    int q = blockIdx.x; long z = blockIdx.y;
    float* row = sc + (z * Nt + q) * (long)NtP;
    __shared__ float red[128];
    int t = threadIdx.x;

    float m = -1e30f;
    for (int j = t; j < Nt; j += 128) m = fmaxf(m, row[j]);
    red[t] = m; __syncthreads();
    for (int o = 64; o > 0; o >>= 1) { if (t < o) red[t] = fmaxf(red[t], red[t + o]); __syncthreads(); }
    float mx = red[0]; __syncthreads();

    float s = 0.f;
    for (int j = t; j < Nt; j += 128) { float e = expf(row[j] - mx); row[j] = e; s += e; }
    red[t] = s; __syncthreads();
    for (int o = 64; o > 0; o >>= 1) { if (t < o) red[t] += red[t + o]; __syncthreads(); }
    float inv = 1.f / red[0];
    for (int j = t; j < Nt; j += 128) row[j] *= inv;
}

// ---------------- exact GELU ----------------
__global__ void gelu_kernel(float* __restrict__ x, long n)
{
    long i = (long)blockIdx.x * blockDim.x + threadIdx.x;
    if (i < n) { float v = x[i]; x[i] = 0.5f * v * (1.f + erff(v * 0.70710678118654752f)); }
}

// ---------------- im2col for 16x16/stride16 patch conv ----------------
__global__ void im2col_kernel(const float* __restrict__ x, float* __restrict__ out)
{
    int bp = blockIdx.x; int b = bp / PNUM, p = bp % PNUM;
    int pr = p / 14, pc = p % 14;
    for (int k = threadIdx.x; k < CDIM; k += blockDim.x) {
        int ci = k >> 8, rem = k & 255, kh = rem >> 4, kw = rem & 15;
        out[((long)b * PNUM + p) * CDIM + k] =
            x[(((long)b * 3 + ci) * 224 + (pr * 16 + kh)) * 224 + (pc * 16 + kw)];
    }
}

// ---------------- assemble tokens: [cls | patches] + pos ----------------
__global__ void assemble_tok_kernel(const float* __restrict__ ptok,
                                    const float* __restrict__ cls,
                                    const float* __restrict__ pos,
                                    float* __restrict__ tok)
{
    int bn = blockIdx.x; int b = bn / NTOK, n = bn % NTOK;
    for (int c = threadIdx.x; c < CDIM; c += blockDim.x) {
        float v = (n == 0) ? cls[c] : ptok[((long)b * PNUM + (n - 1)) * CDIM + c];
        tok[((long)b * NTOK + n) * CDIM + c] = v + pos[(long)n * CDIM + c];
    }
}

// ---------------- MultiHeadVoting ----------------
__global__ void vote_kernel(const float* __restrict__ probs, int Nt, int NtP, int Pn,
                            int select_num, int enhance, int* __restrict__ idx_out)
{
    int b = blockIdx.x;
    __shared__ float cnt[PNUM];
    __shared__ float cnt2[PNUM];
    int t = threadIdx.x;
    for (int j = t; j < Pn; j += blockDim.x) cnt[j] = 0.f;
    __syncthreads();

    if (t < NHEAD) {
        const float* row = probs + (long)(b * NHEAD + t) * Nt * NtP + 1;
        int chosen[NVOTE];
        for (int v = 0; v < NVOTE; ++v) {
            float best = -1e30f; int bi = 0;
            for (int j = 0; j < Pn; ++j) {
                bool used = false;
                for (int u = 0; u < v; ++u) if (chosen[u] == j) { used = true; break; }
                if (used) continue;
                float sv = row[j];
                if (sv > best) { best = sv; bi = j; }
            }
            chosen[v] = bi;
            atomicAdd(&cnt[bi], 1.0f);
        }
    }
    __syncthreads();

    if (enhance) {
        for (int j = t; j < Pn; j += blockDim.x) {
            int r = j / 14, c = j % 14;
            float acc = 0.f;
            for (int dr = -1; dr <= 1; ++dr)
                for (int dc = -1; dc <= 1; ++dc) {
                    int rr = r + dr, cc = c + dc;
                    if (rr < 0 || rr >= 14 || cc < 0 || cc >= 14) continue;
                    float wgt = (float)((dr == 0 ? 2 : 1) * (dc == 0 ? 2 : 1));
                    acc += wgt * cnt[rr * 14 + cc];
                }
            cnt2[j] = acc;
        }
    } else {
        for (int j = t; j < Pn; j += blockDim.x) cnt2[j] = cnt[j];
    }
    __syncthreads();

    if (t == 0) {
        for (int s = 0; s < select_num; ++s) {
            float best = -1e30f; int bi = 0;
            for (int j = 0; j < Pn; ++j) {
                if (cnt2[j] > best) { best = cnt2[j]; bi = j; }
            }
            idx_out[b * select_num + s] = bi + 1;
            cnt2[bi] = -1e31f;
        }
    }
}

// ---------------- gather / copy rows ----------------
__global__ void gather_rows_kernel(const float* __restrict__ src, long sSb,
                                   float* __restrict__ dst, long sDb, int dRow0,
                                   const int* __restrict__ idx, int S)
{
    int b = blockIdx.y, s = blockIdx.x;
    int r = idx[b * S + s];
    const float* in  = src + (long)b * sSb + (long)r * CDIM;
    float*       out = dst + (long)b * sDb + (long)(dRow0 + s) * CDIM;
    for (int c = threadIdx.x; c < CDIM; c += blockDim.x) out[c] = in[c];
}

__global__ void copy_row_kernel(const float* __restrict__ src, long sSb, int srow,
                                float* __restrict__ dst, long sDb, int drow)
{
    int b = blockIdx.x;
    for (int c = threadIdx.x; c < CDIM; c += blockDim.x)
        dst[(long)b * sDb + (long)drow * CDIM + c] =
        src[(long)b * sSb + (long)srow * CDIM + c];
}

// ---------------- classifier head ----------------
__global__ void head_kernel(const float* __restrict__ xv, long sxb,
                            const float* __restrict__ hw, const float* __restrict__ hb,
                            float* __restrict__ out, long sob)
{
    int b = blockIdx.y;
    int c = blockIdx.x * 64 + threadIdx.x;
    if (c >= NCLS) return;
    const float* xr = xv + (long)b * sxb;
    float acc = hb[c];
    for (int j = 0; j < CDIM; ++j) acc += xr[j] * hw[(long)c * CDIM + j];
    out[(long)b * sob + c] = acc;
}

// ---------------- assist: out += softmax(comp_logits) * head_w.sum(-1) ----------------
__global__ void assist_kernel(const float* __restrict__ cl,
                              const float* __restrict__ hw,
                              float* __restrict__ out)
{
    int b = blockIdx.x;
    __shared__ float red[256];
    int t = threadIdx.x;

    float m = -1e30f;
    for (int c = t; c < NCLS; c += 256) m = fmaxf(m, cl[b * NCLS + c]);
    red[t] = m; __syncthreads();
    for (int o = 128; o > 0; o >>= 1) { if (t < o) red[t] = fmaxf(red[t], red[t + o]); __syncthreads(); }
    float mx = red[0]; __syncthreads();

    float s = 0.f;
    for (int c = t; c < NCLS; c += 256) s += expf(cl[b * NCLS + c] - mx);
    red[t] = s; __syncthreads();
    for (int o = 128; o > 0; o >>= 1) { if (t < o) red[t] += red[t + o]; __syncthreads(); }
    float tot = red[0];

    for (int c = t; c < NCLS; c += 256) {
        float rs = 0.f;
        for (int j = 0; j < CDIM; ++j) rs += hw[(long)c * CDIM + j];
        out[b * NCLS + c] += expf(cl[b * NCLS + c] - mx) / tot * rs;
    }
}

// =====================================================================
// Host-side orchestration
// =====================================================================
static void gemm(hipStream_t st, const float* A, long sAb, long sAh, int lda,
                 const float* W, long sWb, long sWh, int ldw, int wtrans,
                 float* C, long sCb, long sCh, int ldc,
                 const float* bias, const float* R, long sRb, int ldr,
                 int M, int N, int K, float alpha, int H)
{
    dim3 g((N + 31) / 32, (M + 15) / 16, BSZ * H);
    gemm16_wmma<<<g, 32, 0, st>>>(A, sAb, sAh, lda, W, sWb, sWh, ldw, wtrans,
                                  C, sCb, sCh, ldc, bias, R, sRb, ldr,
                                  M, N, K, alpha, H);
}

struct BlockParams {
    const float *ln1_s, *ln1_b, *ln2_s, *ln2_b;
    const float *wq, *bq, *wk, *bk, *wv, *bv, *wo, *bo;
    const float *w1, *b1, *w2, *b2;
};

static void run_block(hipStream_t st, int i, int Nt,
                      float* tin, float* tmid, float* tout,
                      float* yb, float* qb, float* kb, float* vb,
                      float* sc, float* ctx, float* h1, const BlockParams& P)
{
    const int NtP = (Nt + 3) & ~3;          // padded scores row stride (vec4 alignment)
    long sT  = (long)Nt * CDIM;
    long sH1 = (long)Nt * MLPD;
    long sS1 = (long)Nt * NtP;
    long sS  = (long)NHEAD * sS1;
    dim3 lng(Nt, BSZ);

    ln_kernel<<<lng, 256, 0, st>>>(tin, sT, P.ln1_s + (long)i * CDIM, P.ln1_b + (long)i * CDIM, yb, sT);

    const long wOff = (long)i * CDIM * CDIM;
    gemm(st, yb, sT, 0, CDIM, P.wq + wOff, 0, 0, CDIM, 0,
         qb, sT, 0, CDIM, P.bq + (long)i * CDIM, nullptr, 0, 0, Nt, CDIM, CDIM, 1.f, 1);
    gemm(st, yb, sT, 0, CDIM, P.wk + wOff, 0, 0, CDIM, 0,
         kb, sT, 0, CDIM, P.bk + (long)i * CDIM, nullptr, 0, 0, Nt, CDIM, CDIM, 1.f, 1);
    gemm(st, yb, sT, 0, CDIM, P.wv + wOff, 0, 0, CDIM, 0,
         vb, sT, 0, CDIM, P.bv + (long)i * CDIM, nullptr, 0, 0, Nt, CDIM, CDIM, 1.f, 1);

    // scores[b,h] = (Q_h @ K_h^T) / sqrt(64)   (row stride NtP)
    gemm(st, qb, sT, DH, CDIM, kb, sT, DH, CDIM, 1,
         sc, sS, sS1, NtP, nullptr, nullptr, 0, 0, Nt, Nt, DH, 0.125f, NHEAD);
    softmax_kernel<<<dim3(Nt, BSZ * NHEAD), 128, 0, st>>>(sc, Nt, NtP);
    // ctx[b,:,h*64:] = probs @ V_h
    gemm(st, sc, sS, sS1, NtP, vb, sT, DH, CDIM, 0,
         ctx, sT, DH, CDIM, nullptr, nullptr, 0, 0, Nt, DH, Nt, 1.f, NHEAD);

    gemm(st, ctx, sT, 0, CDIM, P.wo + wOff, 0, 0, CDIM, 0,
         tmid, sT, 0, CDIM, P.bo + (long)i * CDIM, tin, sT, CDIM, Nt, CDIM, CDIM, 1.f, 1);

    ln_kernel<<<lng, 256, 0, st>>>(tmid, sT, P.ln2_s + (long)i * CDIM, P.ln2_b + (long)i * CDIM, yb, sT);

    gemm(st, yb, sT, 0, CDIM, P.w1 + (long)i * CDIM * MLPD, 0, 0, MLPD, 0,
         h1, sH1, 0, MLPD, P.b1 + (long)i * MLPD, nullptr, 0, 0, Nt, MLPD, CDIM, 1.f, 1);
    long ne = (long)BSZ * Nt * MLPD;
    gelu_kernel<<<(ne + 255) / 256, 256, 0, st>>>(h1, ne);
    gemm(st, h1, sH1, 0, MLPD, P.w2 + (long)i * MLPD * CDIM, 0, 0, CDIM, 0,
         tout, sT, 0, CDIM, P.b2 + (long)i * CDIM, tmid, sT, CDIM, Nt, CDIM, MLPD, 1.f, 1);
}

extern "C" void kernel_launch(void* const* d_in, const int* in_sizes, int n_in,
                              void* d_out, int out_size, void* d_ws, size_t ws_size,
                              hipStream_t stream)
{
    const float* x        = (const float*)d_in[0];
    const float* patch_w  = (const float*)d_in[1];
    const float* patch_b  = (const float*)d_in[2];
    const float* clsv     = (const float*)d_in[3];
    const float* pos      = (const float*)d_in[4];
    BlockParams P;
    P.ln1_s = (const float*)d_in[5];  P.ln1_b = (const float*)d_in[6];
    P.ln2_s = (const float*)d_in[7];  P.ln2_b = (const float*)d_in[8];
    P.wq = (const float*)d_in[9];  P.bq = (const float*)d_in[10];
    P.wk = (const float*)d_in[11]; P.bk = (const float*)d_in[12];
    P.wv = (const float*)d_in[13]; P.bv = (const float*)d_in[14];
    P.wo = (const float*)d_in[15]; P.bo = (const float*)d_in[16];
    P.w1 = (const float*)d_in[17]; P.b1 = (const float*)d_in[18];
    P.w2 = (const float*)d_in[19]; P.b2 = (const float*)d_in[20];
    const float* clr_ln_s = (const float*)d_in[21];
    const float* clr_ln_b = (const float*)d_in[22];
    const float* key_ln_s = (const float*)d_in[23];
    const float* key_ln_b = (const float*)d_in[24];
    const float* head_w   = (const float*)d_in[25];
    const float* head_b   = (const float*)d_in[26];
    float* out = (float*)d_out;

    // ---- carve workspace (all offsets stay 16B aligned) ----
    float* wsp = (float*)d_ws;
    auto carve = [&](long n) { float* p = wsp; wsp += n; return p; };
    float* im2    = carve((long)BSZ * PNUM * CDIM);
    float* ptok   = carve((long)BSZ * PNUM * CDIM);
    float* t0     = carve((long)BSZ * NTOK * CDIM);
    float* t1     = carve((long)BSZ * NTOK * CDIM);
    float* yb     = carve((long)BSZ * NTOK * CDIM);
    float* qb     = carve((long)BSZ * NTOK * CDIM);
    float* kb     = carve((long)BSZ * NTOK * CDIM);
    float* vb     = carve((long)BSZ * NTOK * CDIM);
    float* ctx    = carve((long)BSZ * NTOK * CDIM);
    float* sc     = carve((long)BSZ * NHEAD * NTOK * 200);   // padded stride (NtP<=200)
    float* h1     = carve((long)BSZ * NTOK * MLPD);
    float* clr_in = carve((long)BSZ * 127 * CDIM);
    float* bx     = carve((long)BSZ * 127 * CDIM);
    float* clrb   = carve((long)BSZ * 127 * CDIM);
    float* key_in = carve((long)BSZ * 25 * CDIM);
    float* by     = carve((long)BSZ * 25 * CDIM);
    float* keyb   = carve((long)BSZ * 25 * CDIM);
    float* logc   = carve((long)BSZ * NCLS);
    int*   idxb   = (int*)carve((long)BSZ * NVOTE);

    // ---- patch embedding as WMMA GEMM ----
    im2col_kernel<<<BSZ * PNUM, 256, 0, stream>>>(x, im2);
    gemm(stream, im2, (long)PNUM * CDIM, 0, CDIM, patch_w, 0, 0, CDIM, 1,
         ptok, (long)PNUM * CDIM, 0, CDIM, patch_b, nullptr, 0, 0,
         PNUM, CDIM, CDIM, 1.f, 1);
    assemble_tok_kernel<<<BSZ * NTOK, 256, 0, stream>>>(ptok, clsv, pos, t0);

    // ---- 11 blocks with voting + token compaction ----
    static const int SEL[11] = {16, 14, 12, 10, 8, 6, 8, 10, 12, 14, 16};
    int cum = 1;
    for (int t = 0; t < 11; ++t) {
        run_block(stream, t, NTOK, t0, t1, t0, yb, qb, kb, vb, sc, ctx, h1, P);
        vote_kernel<<<BSZ, 64, 0, stream>>>(sc, NTOK, 200, PNUM, SEL[t], 1, idxb);
        gather_rows_kernel<<<dim3(SEL[t], BSZ), 256, 0, stream>>>(
            t0, (long)NTOK * CDIM, clr_in, (long)127 * CDIM, cum, idxb, SEL[t]);
        cum += SEL[t];
    }

    // ---- clr branch (block 11, N=127, NtP=128) ----
    copy_row_kernel<<<BSZ, 256, 0, stream>>>(t0, (long)NTOK * CDIM, 0,
                                             clr_in, (long)127 * CDIM, 0);
    run_block(stream, 11, 127, clr_in, t1, bx, yb, qb, kb, vb, sc, ctx, h1, P);
    ln_kernel<<<dim3(127, BSZ), 256, 0, stream>>>(bx, (long)127 * CDIM,
                                                  clr_ln_s, clr_ln_b,
                                                  clrb, (long)127 * CDIM);
    vote_kernel<<<BSZ, 64, 0, stream>>>(sc, 127, 128, 126, NVOTE, 0, idxb);

    // ---- key branch (block 12, N=25) ----
    copy_row_kernel<<<BSZ, 256, 0, stream>>>(t0, (long)NTOK * CDIM, 0,
                                             key_in, (long)25 * CDIM, 0);
    gather_rows_kernel<<<dim3(NVOTE, BSZ), 256, 0, stream>>>(
        clrb, (long)127 * CDIM, key_in, (long)25 * CDIM, 1, idxb, NVOTE);
    run_block(stream, 12, 25, key_in, t1, by, yb, qb, kb, vb, sc, ctx, h1, P);
    ln_kernel<<<dim3(25, BSZ), 256, 0, stream>>>(by, (long)25 * CDIM,
                                                 key_ln_s, key_ln_b,
                                                 keyb, (long)25 * CDIM);

    // ---- heads + assist ----
    head_kernel<<<dim3(4, BSZ), 64, 0, stream>>>(clrb, (long)127 * CDIM, head_w, head_b, logc, NCLS);
    head_kernel<<<dim3(4, BSZ), 64, 0, stream>>>(keyb, (long)25 * CDIM, head_w, head_b, out, NCLS);
    assist_kernel<<<BSZ, 256, 0, stream>>>(logc, head_w, out);
}
